// TransformerDecoderLayer_62380105008263
// MI455X (gfx1250) — matrix operand order repaired and checked by reading, not compiled, not run on hardware
//
#include <hip/hip_runtime.h>
#include <hip/hip_bf16.h>
#include <stdint.h>

// ---------------------------------------------------------------------------
// CDNA5 (gfx1250) transformer decoder layer.
// GEMM-shaped work runs on v_wmma_f32_16x16x32_f16; global->LDS staging uses
// the gfx1250 async-to-LDS path (ASYNCcnt) with double buffering; softmax row
// reductions use DPP16 butterflies instead of LDS bpermute.
// ---------------------------------------------------------------------------

typedef __attribute__((ext_vector_type(16))) _Float16 v16h;
typedef __attribute__((ext_vector_type(8)))  float    v8f;

union FragU { v16h h; uint32_t u[8]; };

// ---- async global->LDS copy (GLOBAL_LOAD_ASYNC_TO_LDS_B128) ----------------
#if __has_builtin(__builtin_amdgcn_global_load_async_to_lds_b128)
#define HAVE_ASYNC_LDS 1
typedef int v4i_vs __attribute__((vector_size(16)));
__device__ __forceinline__ void async_cp16(const void* g, void* l) {
  __builtin_amdgcn_global_load_async_to_lds_b128(
      (__attribute__((address_space(1))) v4i_vs*)(void*)(uintptr_t)(g),
      (__attribute__((address_space(3))) v4i_vs*)(l), 0, 0);
}
__device__ __forceinline__ void wait_async0() {
#if __has_builtin(__builtin_amdgcn_s_wait_asynccnt)
  __builtin_amdgcn_s_wait_asynccnt(0);
#else
  asm volatile("s_wait_asynccnt 0" ::: "memory");
#endif
}
#else
#define HAVE_ASYNC_LDS 0
__device__ __forceinline__ void wait_async0() {}
#endif

// ---- DPP16 butterfly reductions (no LDS traffic) ---------------------------
#if __has_builtin(__builtin_amdgcn_update_dpp)
#define HAVE_DPP 1
template <int CTRL>
__device__ __forceinline__ float dpp_f(float x) {
  return __builtin_bit_cast(
      float, __builtin_amdgcn_update_dpp(0, __builtin_bit_cast(int, x),
                                         CTRL, 0xF, 0xF, true));
}
// 16-lane allreduce: xor1, xor2, (8-group partner), (16-group partner)
__device__ __forceinline__ float red16_max(float x) {
  x = fmaxf(x, dpp_f<0xB1>(x));    // quad_perm(1,0,3,2)
  x = fmaxf(x, dpp_f<0x4E>(x));    // quad_perm(2,3,0,1)
  x = fmaxf(x, dpp_f<0x141>(x));   // row_half_mirror
  x = fmaxf(x, dpp_f<0x140>(x));   // row_mirror
  return x;
}
__device__ __forceinline__ float red16_sum(float x) {
  x += dpp_f<0xB1>(x);
  x += dpp_f<0x4E>(x);
  x += dpp_f<0x141>(x);
  x += dpp_f<0x140>(x);
  return x;
}
#else
#define HAVE_DPP 0
__device__ __forceinline__ float red16_max(float x) {
#pragma unroll
  for (int off = 1; off <= 8; off <<= 1) x = fmaxf(x, __shfl_xor(x, off, 32));
  return x;
}
__device__ __forceinline__ float red16_sum(float x) {
#pragma unroll
  for (int off = 1; off <= 8; off <<= 1) x += __shfl_xor(x, off, 32);
  return x;
}
#endif

__device__ __forceinline__ float fast_exp2(float x) {
#if __has_builtin(__builtin_amdgcn_exp2f)
  return __builtin_amdgcn_exp2f(x);   // raw v_exp_f32 (FTZ ok for softmax)
#else
  return exp2f(x);
#endif
}

// ---- WMMA fragment loaders (contiguous k = 16*(l/16) + s, same in A and B) ------------------
// 16-bit A matrix 16x32: lane l holds row (l&15); lanes 0-15: K=0..15,
// lanes 16-31: K=16..31; 2 consecutive K per dword.
__device__ __forceinline__ v16h load_frag_A(const _Float16* rowptr, int kbase, int lane) {
  FragU f;
  int kh = kbase + ((lane >> 4) << 4);
#pragma unroll
  for (int v = 0; v < 8; ++v)
    f.u[v] = *(const uint32_t*)(rowptr + kh + (v << 1));
  return f.h;
}

// 16-bit B matrix 32x16: lane l holds column (l&15); lanes 0-15: K=0..15,
// lanes 16-31: K=16..31; dword v holds K = {2v, 2v+1}. rowptr = row-major
// [N][K] row n (exactly how W (out,in) stores data for x @ W.T).
__device__ __forceinline__ v16h load_frag_B(const _Float16* rowptr, int kbase, int lane) {
  FragU f;
  int kh = kbase + ((lane >> 4) << 4);
#pragma unroll
  for (int v = 0; v < 8; ++v)
    f.u[v] = *(const uint32_t*)(rowptr + kh + (v << 1));
  return f.h;
}

// B operand from [K][N] storage (V matrix): strided 16-bit gather + pack.
__device__ __forceinline__ v16h load_frag_B_strided(const _Float16* colptr, int ld, int lane) {
  FragU f;
  int k0 = (lane >> 4) << 4;
#pragma unroll
  for (int v = 0; v < 8; ++v) {
    uint32_t lo = *(const uint16_t*)(colptr + (size_t)(k0 + 2 * v) * ld);
    uint32_t hi = *(const uint16_t*)(colptr + (size_t)(k0 + 2 * v + 1) * ld);
    f.u[v] = lo | (hi << 16);
  }
  return f.h;
}

__device__ __forceinline__ v8f wmma_f16(v16h a, v16h b, v8f c) {
  return __builtin_amdgcn_wmma_f32_16x16x32_f16(false, a, false, b, (short)0, c, false, false);
}

// ---------------------------------------------------------------------------
// Tiled WMMA GEMM:  Y[M,N] = X[M,K] @ W[N,K]^T + bias  (X,W f16; acc f32)
// block = 128x128, 256 threads (8 waves in 4x2), wave tile = 32x64
// Double-buffered LDS; async DMA of tile i+1 overlaps WMMA of tile i.
// MODE 0: f16 out   MODE 1: f16 relu out   MODE 2: f32 out += acc + bias
// ---------------------------------------------------------------------------
#define LDT 40  // LDS row stride in halves (32 + 8 pad)

template <int MODE>
__global__ __launch_bounds__(256) void gemm_wmma_k(
    const _Float16* __restrict__ A, const _Float16* __restrict__ W,
    const float* __restrict__ bias, void* __restrict__ out,
    int M, int N, int K) {
  __shared__ _Float16 At[2][128 * LDT];
  __shared__ _Float16 Bt[2][128 * LDT];

  const int tid  = threadIdx.x;
  const int lane = tid & 31;
  const int wave = tid >> 5;
  const int wm = wave >> 1;   // 0..3
  const int wn = wave & 1;    // 0..1
  const int bm = blockIdx.y;
  const int bn = blockIdx.x;

  v8f acc[2][4];
#pragma unroll
  for (int mt = 0; mt < 2; ++mt)
#pragma unroll
    for (int nt = 0; nt < 4; ++nt) acc[mt][nt] = {};

  auto stage = [&](int buf, int k0) {
#if HAVE_ASYNC_LDS
#pragma unroll
    for (int i = 0; i < 2; ++i) {
      int e = tid + i * 256, row = e >> 2, c = e & 3;  // 4 x 16B per 32-half row
      async_cp16(A + (size_t)(bm * 128 + row) * K + k0 + c * 8,
                 At[buf] + row * LDT + c * 8);
      async_cp16(W + (size_t)(bn * 128 + row) * K + k0 + c * 8,
                 Bt[buf] + row * LDT + c * 8);
    }
#else
    uint4 ra[2], rb[2];
#pragma unroll
    for (int i = 0; i < 2; ++i) {
      int e = tid + i * 256, row = e >> 2, c = e & 3;
      ra[i] = *(const uint4*)(A + (size_t)(bm * 128 + row) * K + k0 + c * 8);
      rb[i] = *(const uint4*)(W + (size_t)(bn * 128 + row) * K + k0 + c * 8);
    }
#pragma unroll
    for (int i = 0; i < 2; ++i) {
      int e = tid + i * 256, row = e >> 2, c = e & 3;
      *(uint4*)(At[buf] + row * LDT + c * 8) = ra[i];
      *(uint4*)(Bt[buf] + row * LDT + c * 8) = rb[i];
    }
#endif
  };

  const int nk = K >> 5;
  stage(0, 0);

  for (int i = 0; i < nk; ++i) {
    const int buf = i & 1;
    wait_async0();       // per-wave: its copies of tile i are in LDS
    __syncthreads();     // all waves staged tile i / done reading tile i-1
    if (i + 1 < nk) stage(buf ^ 1, (i + 1) << 5);

    v16h af[2], bf[4];
#pragma unroll
    for (int mt = 0; mt < 2; ++mt)
      af[mt] = load_frag_A(At[buf] + (wm * 32 + mt * 16 + (lane & 15)) * LDT, 0, lane);
#pragma unroll
    for (int nt = 0; nt < 4; ++nt)
      bf[nt] = load_frag_B(Bt[buf] + (wn * 64 + nt * 16 + (lane & 15)) * LDT, 0, lane);
#pragma unroll
    for (int mt = 0; mt < 2; ++mt)
#pragma unroll
      for (int nt = 0; nt < 4; ++nt)
        acc[mt][nt] = wmma_f16(af[mt], bf[nt], acc[mt][nt]);
  }

  // epilogue: C layout = 8 VGPRs, lane holds col (lane&15), rows (lane>>4)*8+r
#pragma unroll
  for (int mt = 0; mt < 2; ++mt) {
#pragma unroll
    for (int nt = 0; nt < 4; ++nt) {
      int gm0 = bm * 128 + wm * 32 + mt * 16 + ((lane >> 4) << 3);
      int gn  = bn * 128 + wn * 64 + nt * 16 + (lane & 15);
      float bb = bias[gn];
#pragma unroll
      for (int r = 0; r < 8; ++r) {
        float v = acc[mt][nt][r] + bb;
        size_t idx = (size_t)(gm0 + r) * N + gn;
        if (MODE == 0) ((_Float16*)out)[idx] = (_Float16)v;
        if (MODE == 1) ((_Float16*)out)[idx] = (_Float16)fmaxf(v, 0.0f);
        if (MODE == 2) ((float*)out)[idx] += v;
      }
    }
  }
}

// ---------------------------------------------------------------------------
// Flash attention (hd = 64, 16 heads). Grid: (B*16, Lq/64), 128 threads.
// Each wave owns 16 q rows; K/V staged in LDS 32 rows at a time (async DMA).
// ---------------------------------------------------------------------------
__global__ __launch_bounds__(128) void attn_wmma_k(
    const _Float16* __restrict__ Q, const _Float16* __restrict__ Kb,
    const _Float16* __restrict__ Vb, _Float16* __restrict__ O,
    int Lq, int Lk, int D) {
  __shared__ _Float16 Kt[32 * 72];
  __shared__ _Float16 Vt[32 * 72];
  __shared__ _Float16 Pt[4][16 * 32];

  const int tid  = threadIdx.x;
  const int lane = tid & 31;
  const int wave = tid >> 5;
  const int b = blockIdx.x >> 4;
  const int h = blockIdx.x & 15;
  const int qBase = blockIdx.y * 64;

  const float cscale = 0.125f * 1.44269504088896f;  // 1/sqrt(64) * log2(e)

  // Q fragments for this wave's 16 rows (hd=64 -> 2 K-slices of 32)
  const _Float16* qrow =
      Q + ((size_t)b * Lq + qBase + wave * 16 + (lane & 15)) * D + h * 64;
  v16h qf0 = load_frag_A(qrow, 0, lane);
  v16h qf1 = load_frag_A(qrow, 32, lane);

  v8f o[4];
#pragma unroll
  for (int nt = 0; nt < 4; ++nt) o[nt] = {};
  float m[8], l[8];
#pragma unroll
  for (int r = 0; r < 8; ++r) { m[r] = -INFINITY; l[r] = 0.0f; }

  for (int kt0 = 0; kt0 < Lk; kt0 += 32) {
    __syncthreads();  // everyone done reading Kt/Vt/Pt from previous tile
    // stage K,V tiles (32 x 64 halves each): 2 x 16B per thread per tile
#if HAVE_ASYNC_LDS
#pragma unroll
    for (int i = 0; i < 2; ++i) {
      int e = tid + i * 128, row = e >> 3, c = e & 7;
      size_t src = ((size_t)b * Lk + kt0 + row) * D + h * 64 + c * 8;
      async_cp16(Kb + src, Kt + row * 72 + c * 8);
      async_cp16(Vb + src, Vt + row * 72 + c * 8);
    }
    wait_async0();
#else
    {
      uint4 rk[2], rv[2];
#pragma unroll
      for (int i = 0; i < 2; ++i) {
        int e = tid + i * 128, row = e >> 3, c = e & 7;
        size_t src = ((size_t)b * Lk + kt0 + row) * D + h * 64 + c * 8;
        rk[i] = *(const uint4*)(Kb + src);
        rv[i] = *(const uint4*)(Vb + src);
      }
#pragma unroll
      for (int i = 0; i < 2; ++i) {
        int e = tid + i * 128, row = e >> 3, c = e & 7;
        *(uint4*)(Kt + row * 72 + c * 8) = rk[i];
        *(uint4*)(Vt + row * 72 + c * 8) = rv[i];
      }
    }
#endif
    __syncthreads();

    // S = Q . K^T  (16 x 32 per wave)
    v8f s[2];
#pragma unroll
    for (int nt = 0; nt < 2; ++nt) {
      s[nt] = {};
      const _Float16* krow = Kt + (nt * 16 + (lane & 15)) * 72;
      s[nt] = wmma_f16(qf0, load_frag_B(krow, 0, lane), s[nt]);
      s[nt] = wmma_f16(qf1, load_frag_B(krow, 32, lane), s[nt]);
    }

    // online softmax; row lives in the lane's 16-lane half -> DPP reductions
#pragma unroll
    for (int r = 0; r < 8; ++r) {
      float t0 = s[0][r] * cscale;
      float t1 = s[1][r] * cscale;
      float mx = red16_max(fmaxf(t0, t1));
      float mN = fmaxf(m[r], mx);
      float p0 = fast_exp2(t0 - mN);
      float p1 = fast_exp2(t1 - mN);
      float alpha = fast_exp2(m[r] - mN);
      float rs = red16_sum(p0 + p1);
      l[r] = l[r] * alpha + rs;
      m[r] = mN;
#pragma unroll
      for (int nt = 0; nt < 4; ++nt) o[nt][r] *= alpha;
      int prow = ((lane >> 4) << 3) + r;
      Pt[wave][prow * 32 + (lane & 15)]      = (_Float16)p0;
      Pt[wave][prow * 32 + 16 + (lane & 15)] = (_Float16)p1;
    }
    __syncthreads();

    // O += P . V   (P: 16x32 A-frag from LDS; V: [32][64] strided B operand)
    v16h pa = load_frag_A(Pt[wave] + (lane & 15) * 32, 0, lane);
#pragma unroll
    for (int nt = 0; nt < 4; ++nt) {
      v16h vb = load_frag_B_strided(Vt + nt * 16 + (lane & 15), 72, lane);
      o[nt] = wmma_f16(pa, vb, o[nt]);
    }
  }

  // normalize and write out (f16), layout [b, q, h*64+d]
#pragma unroll
  for (int r = 0; r < 8; ++r) {
    float inv = 1.0f / l[r];
    int orow = qBase + wave * 16 + ((lane >> 4) << 3) + r;
#pragma unroll
    for (int nt = 0; nt < 4; ++nt) {
      int ocol = h * 64 + nt * 16 + (lane & 15);
      O[((size_t)b * Lq + orow) * D + ocol] = (_Float16)(o[nt][r] * inv);
    }
  }
}

// ---------------------------------------------------------------------------
// Elementwise helpers
// ---------------------------------------------------------------------------
__global__ void f32_to_f16_k(const float* __restrict__ in,
                             _Float16* __restrict__ out, size_t n) {
  size_t i = (size_t)blockIdx.x * blockDim.x + threadIdx.x;
  size_t stride = (size_t)gridDim.x * blockDim.x;
  for (; i < n; i += stride) out[i] = (_Float16)in[i];
}

__global__ void copy_f32_k(const float* __restrict__ in,
                           float* __restrict__ out, size_t n) {
  size_t i = (size_t)blockIdx.x * blockDim.x + threadIdx.x;
  size_t stride = (size_t)gridDim.x * blockDim.x;
  for (; i < n; i += stride) out[i] = in[i];
}

// LayerNorm over D=1024, one row per 256-thread block, f32 in -> f16 out
__global__ __launch_bounds__(256) void layernorm_f16_k(
    const float* __restrict__ x, const float* __restrict__ g,
    const float* __restrict__ bta, _Float16* __restrict__ y) {
  __shared__ float red[18];
  const int tid = threadIdx.x, lane = tid & 31, wid = tid >> 5;
  const float* xr = x + (size_t)blockIdx.x * 1024;
  float s = 0.0f, ss = 0.0f;
#pragma unroll
  for (int i = tid; i < 1024; i += 256) {
    float v = xr[i];
    s += v; ss += v * v;
  }
  // 16-lane DPP allreduce, then fold halves with one shfl, then cross-wave
  s  = red16_sum(s);
  ss = red16_sum(ss);
  s  += __shfl_xor(s, 16, 32);
  ss += __shfl_xor(ss, 16, 32);
  if (lane == 0) { red[wid] = s; red[8 + wid] = ss; }
  __syncthreads();
  if (tid == 0) {
    float S = 0.0f, SS = 0.0f;
#pragma unroll
    for (int w = 0; w < 8; ++w) { S += red[w]; SS += red[8 + w]; }
    float mean = S * (1.0f / 1024.0f);
    float var = SS * (1.0f / 1024.0f) - mean * mean;
    red[16] = mean;
    red[17] = rsqrtf(var + 1e-5f);
  }
  __syncthreads();
  float mean = red[16], rstd = red[17];
  _Float16* yr = y + (size_t)blockIdx.x * 1024;
#pragma unroll
  for (int i = tid; i < 1024; i += 256)
    yr[i] = (_Float16)((xr[i] - mean) * rstd * g[i] + bta[i]);
}

// ---------------------------------------------------------------------------
// Host-side orchestration
// ---------------------------------------------------------------------------
extern "C" void kernel_launch(void* const* d_in, const int* in_sizes, int n_in,
                              void* d_out, int out_size, void* d_ws, size_t ws_size,
                              hipStream_t stream) {
  (void)in_sizes; (void)n_in; (void)out_size; (void)ws_size;

  const int Bb = 8, Lq = 1024, Lk = 1024, D = 1024, Hf = 4096;
  const int Mq = Bb * Lq;   // 8192 rows for q-side GEMMs
  const int Mk = Bb * Lk;   // 8192 rows for mem-side GEMMs

  const float* x    = (const float*)d_in[0];
  const float* mem  = (const float*)d_in[1];
  const float* ln1g = (const float*)d_in[2];
  const float* ln1b = (const float*)d_in[3];
  const float* saWq = (const float*)d_in[4];  const float* sabq = (const float*)d_in[5];
  const float* saWk = (const float*)d_in[6];  const float* sabk = (const float*)d_in[7];
  const float* saWv = (const float*)d_in[8];  const float* sabv = (const float*)d_in[9];
  const float* saWo = (const float*)d_in[10]; const float* sabo = (const float*)d_in[11];
  const float* ln2g = (const float*)d_in[12];
  const float* ln2b = (const float*)d_in[13];
  const float* caWq = (const float*)d_in[14]; const float* cabq = (const float*)d_in[15];
  const float* caWk = (const float*)d_in[16]; const float* cabk = (const float*)d_in[17];
  const float* caWv = (const float*)d_in[18]; const float* cabv = (const float*)d_in[19];
  const float* caWo = (const float*)d_in[20]; const float* cabo = (const float*)d_in[21];
  const float* ln3g = (const float*)d_in[22];
  const float* ln3b = (const float*)d_in[23];
  const float* fW1  = (const float*)d_in[24]; const float* fb1 = (const float*)d_in[25];
  const float* fW2  = (const float*)d_in[26]; const float* fb2 = (const float*)d_in[27];

  float* xout = (float*)d_out;          // residual stream lives here (f32)

  // ---- workspace carve-up (128 MB total) ----
  char* ws = (char*)d_ws;
  const size_t MB = 1ull << 20;
  _Float16* wSaQ = (_Float16*)(ws + 0 * MB);
  _Float16* wSaK = (_Float16*)(ws + 2 * MB);
  _Float16* wSaV = (_Float16*)(ws + 4 * MB);
  _Float16* wSaO = (_Float16*)(ws + 6 * MB);
  _Float16* wCaQ = (_Float16*)(ws + 8 * MB);
  _Float16* wCaK = (_Float16*)(ws + 10 * MB);
  _Float16* wCaV = (_Float16*)(ws + 12 * MB);
  _Float16* wCaO = (_Float16*)(ws + 14 * MB);
  _Float16* wF1  = (_Float16*)(ws + 16 * MB);   // 8 MB
  _Float16* wF2  = (_Float16*)(ws + 24 * MB);   // 8 MB
  _Float16* ln16 = (_Float16*)(ws + 32 * MB);   // 16 MB
  _Float16* mem16= (_Float16*)(ws + 48 * MB);   // 16 MB
  _Float16* q16  = (_Float16*)(ws + 64 * MB);   // 16 MB
  _Float16* k16  = (_Float16*)(ws + 80 * MB);   // 16 MB
  _Float16* v16  = (_Float16*)(ws + 96 * MB);   // 16 MB
  _Float16* at16 = (_Float16*)(ws + 112 * MB);  // 16 MB
  _Float16* hid16= (_Float16*)(ws + 64 * MB);   // 64 MB (aliases q/k/v/at, dead by FFN)

  const size_t nDD = (size_t)D * D;
  const size_t nDH = (size_t)D * Hf;
  const size_t nAct = (size_t)Mq * D;

  auto cvt = [&](const float* src, _Float16* dst, size_t n) {
    f32_to_f16_k<<<2048, 256, 0, stream>>>(src, dst, n);
  };
  // weight + activation conversions
  cvt(saWq, wSaQ, nDD); cvt(saWk, wSaK, nDD); cvt(saWv, wSaV, nDD); cvt(saWo, wSaO, nDD);
  cvt(caWq, wCaQ, nDD); cvt(caWk, wCaK, nDD); cvt(caWv, wCaV, nDD); cvt(caWo, wCaO, nDD);
  cvt(fW1, wF1, nDH);   cvt(fW2, wF2, nDH);
  cvt(mem, mem16, (size_t)Mk * D);

  // residual stream init: xout = x
  copy_f32_k<<<2048, 256, 0, stream>>>(x, xout, nAct);

  dim3 gD(D / 128, Mq / 128), gH(Hf / 128, Mq / 128);
  dim3 gAtt(Bb * 16, Lq / 64);

  // ---- self-attention block ----
  layernorm_f16_k<<<Mq, 256, 0, stream>>>(xout, ln1g, ln1b, ln16);
  gemm_wmma_k<0><<<gD, 256, 0, stream>>>(ln16, wSaQ, sabq, q16, Mq, D, D);
  gemm_wmma_k<0><<<gD, 256, 0, stream>>>(ln16, wSaK, sabk, k16, Mq, D, D);
  gemm_wmma_k<0><<<gD, 256, 0, stream>>>(ln16, wSaV, sabv, v16, Mq, D, D);
  attn_wmma_k<<<gAtt, 128, 0, stream>>>(q16, k16, v16, at16, Lq, Lk, D);
  gemm_wmma_k<2><<<gD, 256, 0, stream>>>(at16, wSaO, sabo, xout, Mq, D, D);

  // ---- cross-attention block ----
  layernorm_f16_k<<<Mq, 256, 0, stream>>>(xout, ln2g, ln2b, ln16);
  gemm_wmma_k<0><<<gD, 256, 0, stream>>>(ln16, wCaQ, cabq, q16, Mq, D, D);
  gemm_wmma_k<0><<<dim3(D / 128, Mk / 128), 256, 0, stream>>>(mem16, wCaK, cabk, k16, Mk, D, D);
  gemm_wmma_k<0><<<dim3(D / 128, Mk / 128), 256, 0, stream>>>(mem16, wCaV, cabv, v16, Mk, D, D);
  attn_wmma_k<<<gAtt, 128, 0, stream>>>(q16, k16, v16, at16, Lq, Lk, D);
  gemm_wmma_k<2><<<gD, 256, 0, stream>>>(at16, wCaO, cabo, xout, Mq, D, D);

  // ---- FFN block ----
  layernorm_f16_k<<<Mq, 256, 0, stream>>>(xout, ln3g, ln3b, ln16);
  gemm_wmma_k<1><<<gH, 256, 0, stream>>>(ln16, wF1, fb1, hid16, Mq, Hf, D);
  gemm_wmma_k<2><<<gD, 256, 0, stream>>>(hid16, wF2, fb2, xout, Mq, D, Hf);
}
